// SAGpool_50981261804239
// MI455X (gfx1250) — matrix operand order repaired and checked by reading, not compiled, not run on hardware
//
#include <hip/hip_runtime.h>
#include <hip/hip_bf16.h>
#include <math.h>

// ---------------- problem constants (match reference) ----------------
constexpr int cB    = 128;                 // graphs
constexpr int cN1   = 256;                 // nodes per partition per graph
constexpr int cNPG  = 512;                 // nodes per graph
constexpr int cN    = cB * cNPG;           // 65536 nodes
constexpr int cE    = 1048576;             // edges
constexpr int cH    = 128;                 // NHID
constexpr int cD    = 384;                 // concat width
constexpr int cK    = 128;                 // kept nodes per graph-partition
constexpr int cNP   = 2 * cB * cK;         // 32768 pooled nodes (both partitions)
constexpr int cNG   = 2 * cB;              // 256 pooled graphs

// ---------------- WMMA types ----------------
typedef __bf16 bf16_t;
typedef bf16_t v16bf __attribute__((ext_vector_type(16)));
typedef float  v8f   __attribute__((ext_vector_type(8)));

// =====================================================================
// Generic fill
// =====================================================================
__global__ void fill_zero(float* __restrict__ p, int n) {
    int i = blockIdx.x * blockDim.x + threadIdx.x;
    if (i < n) p[i] = 0.0f;
}

// batch_ output: out[i] = i >> 8 for i in [0, 65536)
__global__ void write_batch(float* __restrict__ out) {
    int i = blockIdx.x * blockDim.x + threadIdx.x;
    if (i < cN) out[i] = (float)(i >> 8);
}

// =====================================================================
// WMMA GEMM: C[M x 128] = A[M x Kdim] * B[Kdim x 128]
//
// Block = 256 threads = 8 waves; block computes a 128-row x 128-col tile.
// B (Kdim x 128) is staged once per block into LDS in WMMA-fragment order:
//   fragment f = (k0/32)*8 + nt  holds the 32x16 B-tile for k-step k0,
//   column tile nt, stored as [lane][e] so the inner loop reads one
//   aligned 32B vector per fragment per lane (ds_load_b128 x2).
// Each wave owns 16 rows and all 8 column tiles: 8 accumulators,
// 8 v_wmma_f32_16x16x32_bf16 per K-step; A fragment loaded once per
// K-step as 4x float4 (global_load_b128) and reused by all 8 WMMAs.
// =====================================================================
__global__ void wmma_gemm(const float* __restrict__ A, int lda,
                          const float* __restrict__ Bm,   // Kdim x 128 row-major
                          float* __restrict__ C, int ldc, int Kdim) {
    extern __shared__ __align__(32) bf16_t ldsB[];   // Kdim*128 bf16

    const int tid = threadIdx.x;

    // ---- cooperative stage of B into fragment-major LDS layout ----
    for (int idx = tid; idx < Kdim * 128; idx += 256) {
        int k = idx >> 7;          // B row (K index)
        int n = idx & 127;         // B col
        int frag = ((k >> 5) << 3) + (n >> 4);            // (k0/32)*8 + nt
        int lane = (((k >> 3) & 1) << 4) | (n & 15);      // B-operand lane
        int e    = (k & 7) | (((k >> 4) & 1) << 3);       // element within lane
        ldsB[frag * 512 + lane * 16 + e] = (bf16_t)Bm[idx];
    }
    __syncthreads();

    const int lane = tid & 31;
    const int wave = tid >> 5;
    const int m0   = (blockIdx.x * 8 + wave) * 16;

    // A-operand layout (16-bit A 16x32): lane L holds row M=L%16,
    // K-halves [kbA..kbA+7] and [kbA+16..kbA+23], kbA=(L/16)*8.
    const int Ml  = lane & 15;
    const int kbA = (lane >> 4) * 8;

    v8f acc[8] = {};
    const float* Arow = A + (size_t)(m0 + Ml) * (size_t)lda;

    for (int k0 = 0; k0 < Kdim; k0 += 32) {
        float4 a0 = *(const float4*)(Arow + k0 + kbA);
        float4 a1 = *(const float4*)(Arow + k0 + kbA + 4);
        float4 a2 = *(const float4*)(Arow + k0 + kbA + 16);
        float4 a3 = *(const float4*)(Arow + k0 + kbA + 20);
        v16bf a;
        a[0]  = (bf16_t)a0.x; a[1]  = (bf16_t)a0.y; a[2]  = (bf16_t)a0.z; a[3]  = (bf16_t)a0.w;
        a[4]  = (bf16_t)a1.x; a[5]  = (bf16_t)a1.y; a[6]  = (bf16_t)a1.z; a[7]  = (bf16_t)a1.w;
        a[8]  = (bf16_t)a2.x; a[9]  = (bf16_t)a2.y; a[10] = (bf16_t)a2.z; a[11] = (bf16_t)a2.w;
        a[12] = (bf16_t)a3.x; a[13] = (bf16_t)a3.y; a[14] = (bf16_t)a3.z; a[15] = (bf16_t)a3.w;

        const bf16_t* fb = ldsB + ((k0 >> 5) << 3) * 512 + lane * 16;
#pragma unroll
        for (int nt = 0; nt < 8; ++nt) {
            v16bf b = *(const v16bf*)(fb + nt * 512);
            acc[nt] = __builtin_amdgcn_wmma_f32_16x16x32_bf16(
                          false, a, false, b, (short)0, acc[nt], false, false);
        }
    }

    // C/D layout: VGPR i -> row (lane/16)*8 + i, col lane%16
    const int rb = (lane >> 4) * 8;
#pragma unroll
    for (int nt = 0; nt < 8; ++nt) {
        int col = nt * 16 + (lane & 15);
#pragma unroll
        for (int i = 0; i < 8; ++i) {
            C[(size_t)(m0 + rb + i) * (size_t)ldc + col] = acc[nt][i];
        }
    }
}

// =====================================================================
// Degree / rsqrt
// =====================================================================
__global__ void edge_count(const int* __restrict__ dst, float* __restrict__ deg) {
    int e = blockIdx.x * blockDim.x + threadIdx.x;
    if (e < cE) atomicAdd(&deg[dst[e]], 1.0f);
}

__global__ void rsqrt_deg(const float* __restrict__ deg, float* __restrict__ dinv, int n) {
    int i = blockIdx.x * blockDim.x + threadIdx.x;
    if (i < n) dinv[i] = rsqrtf(deg[i] + 1.0f);
}

// =====================================================================
// GCN edge aggregation: one wave per edge, lane = 4 channels (float4)
// agg[dst] += dinv[src]*dinv[dst] * xw[src]
// =====================================================================
__global__ void gcn_agg(const int* __restrict__ src, const int* __restrict__ dst,
                        const float* __restrict__ dinv, const float* __restrict__ xw,
                        float* __restrict__ agg) {
    int e    = blockIdx.x * 8 + (threadIdx.x >> 5);
    int lane = threadIdx.x & 31;
    if (e >= cE) return;
    int s = src[e], d = dst[e];
    float nrm = dinv[s] * dinv[d];
    float4 v = ((const float4*)(xw + (size_t)s * cH))[lane];
    float* ad = agg + (size_t)d * cH + lane * 4;
    atomicAdd(ad + 0, nrm * v.x);
    atomicAdd(ad + 1, nrm * v.y);
    atomicAdd(ad + 2, nrm * v.z);
    atomicAdd(ad + 3, nrm * v.w);
}

// out[n, c] = relu(agg + xw*dinv^2 + bias), row stride ldo
__global__ void gcn_finish(const float* __restrict__ agg, const float* __restrict__ xw,
                           const float* __restrict__ dinv, const float* __restrict__ bias,
                           float* __restrict__ out, int ldo, int n_nodes) {
    int idx = blockIdx.x * blockDim.x + threadIdx.x;
    if (idx >= n_nodes * cH) return;
    int n = idx >> 7, c = idx & 127;
    float di = dinv[n];
    float v = agg[idx] + xw[idx] * di * di + bias[c];
    out[(size_t)n * ldo + c] = fmaxf(v, 0.0f);
}

// =====================================================================
// SAGPool scorer:  yr[n] = h[n].Wr,  ys[n] = h[n].Ws   (one wave per node)
// =====================================================================
__global__ void score_dots(const float* __restrict__ hcat,
                           const float* __restrict__ Wr, const float* __restrict__ Ws,
                           float* __restrict__ yr, float* __restrict__ ys) {
    int n    = blockIdx.x * 8 + (threadIdx.x >> 5);
    int lane = threadIdx.x & 31;
    if (n >= cN) return;
    const float* hr = hcat + (size_t)n * cD;
    float r = 0.0f, s = 0.0f;
    for (int c = lane; c < cD; c += 32) {
        float h = hr[c];
        r += h * Wr[c];
        s += h * Ws[c];
    }
#pragma unroll
    for (int off = 16; off > 0; off >>= 1) {
        r += __shfl_down(r, off, 32);
        s += __shfl_down(s, off, 32);
    }
    if (lane == 0) { yr[n] = r; ys[n] = s; }
}

// scalar edge scatter: sacc[dst] += yr[src] for same-partition edges
__global__ void edge_score(const int* __restrict__ src, const int* __restrict__ dst,
                           const float* __restrict__ yr, float* __restrict__ sacc) {
    int e = blockIdx.x * blockDim.x + threadIdx.x;
    if (e >= cE) return;
    int s = src[e], d = dst[e];
    bool ps = (s & (cNPG - 1)) < cN1;
    bool pd = (d & (cNPG - 1)) < cN1;
    if (ps == pd) atomicAdd(&sacc[d], yr[s]);
}

__global__ void score_finish(const float* __restrict__ sacc, const float* __restrict__ ys,
                             const float* __restrict__ pb, float* __restrict__ score) {
    int n = blockIdx.x * blockDim.x + threadIdx.x;
    if (n < cN) score[n] = tanhf(sacc[n] + ys[n] + pb[0]);
}

// =====================================================================
// Top-K by rank-count (256 scores -> keep 128). Block = one graph-partition.
// keep[n] = pooled id or -1;  inv[pid] = original node id.
// Tie-break (==, lower index) matches jax.lax.top_k.
// =====================================================================
__global__ void topk_kernel(const float* __restrict__ score,
                            int* __restrict__ keep, int* __restrict__ inv) {
    __shared__ float s[256];
    int gb = blockIdx.x;          // p*B + g, 0..255
    int p  = gb >> 7;
    int g  = gb & 127;
    int t  = threadIdx.x;         // 0..255
    int n  = g * cNPG + p * cN1 + t;
    float my = score[n];
    s[t] = my;
    __syncthreads();
    int cnt = 0;
    for (int j = 0; j < 256; ++j) {
        float v = s[j];
        cnt += (v > my) || (v == my && j < t);
    }
    if (cnt < cK) {
        int pid = gb * cK + cnt;
        keep[n] = pid;
        inv[pid] = n;
    } else {
        keep[n] = -1;
    }
}

// xp[pid, c] = hcat[inv[pid], c] * score[inv[pid]]
__global__ void pool_gather(const float* __restrict__ hcat, const float* __restrict__ score,
                            const int* __restrict__ inv, float* __restrict__ xp) {
    int idx = blockIdx.x * blockDim.x + threadIdx.x;
    if (idx >= cNP * cD) return;
    int pid = idx / cD, c = idx - pid * cD;
    int n = inv[pid];
    xp[idx] = hcat[(size_t)n * cD + c] * score[n];
}

// pooled degree: count surviving edges into keep[dst]
__global__ void pooled_deg(const int* __restrict__ src, const int* __restrict__ dst,
                           const int* __restrict__ keep, float* __restrict__ degp) {
    int e = blockIdx.x * blockDim.x + threadIdx.x;
    if (e >= cE) return;
    int s = src[e], d = dst[e];
    bool ps = (s & (cNPG - 1)) < cN1;
    bool pd = (d & (cNPG - 1)) < cN1;
    if (ps != pd) return;
    int r = keep[s], c = keep[d];
    if (r >= 0 && c >= 0) atomicAdd(&degp[c], 1.0f);
}

// pooled aggregation: one wave per original edge, masked
__global__ void pooled_agg(const int* __restrict__ src, const int* __restrict__ dst,
                           const int* __restrict__ keep, const float* __restrict__ dinvp,
                           const float* __restrict__ xwp, float* __restrict__ aggp) {
    int e    = blockIdx.x * 8 + (threadIdx.x >> 5);
    int lane = threadIdx.x & 31;
    if (e >= cE) return;
    int s = src[e], d = dst[e];
    bool ps = (s & (cNPG - 1)) < cN1;
    bool pd = (d & (cNPG - 1)) < cN1;
    if (ps != pd) return;
    int r = keep[s], c = keep[d];
    if (r < 0 || c < 0) return;
    float nrm = dinvp[r] * dinvp[c];
    float4 v = ((const float4*)(xwp + (size_t)r * cH))[lane];
    float* ad = aggp + (size_t)c * cH + lane * 4;
    atomicAdd(ad + 0, nrm * v.x);
    atomicAdd(ad + 1, nrm * v.y);
    atomicAdd(ad + 2, nrm * v.z);
    atomicAdd(ad + 3, nrm * v.w);
}

// =====================================================================
// Attention pooling
// =====================================================================
__global__ void xsum_kernel(const float* __restrict__ h1, float* __restrict__ xsum) {
    int gg = blockIdx.x;       // 0..255
    int c  = threadIdx.x;      // 0..127
    float acc = 0.0f;
    for (int i = 0; i < cK; ++i)
        acc += h1[((size_t)gg * cK + i) * cH + c];
    xsum[gg * cH + c] = acc * (1.0f / (float)cK);
}

__global__ void cvec_kernel(const float* __restrict__ xsum, const float* __restrict__ gpW,
                            float* __restrict__ cvec) {
    int idx = blockIdx.x * blockDim.x + threadIdx.x;   // gg*128 + c
    if (idx >= cNG * cH) return;
    int gg = idx >> 7, c = idx & 127;
    const float* xs = xsum + gg * cH;
    float acc = 0.0f;
    for (int i = 0; i < cH; ++i) acc += xs[i] * gpW[i * cH + c];
    cvec[idx] = tanhf(acc);
}

__global__ void alpha_kernel(const float* __restrict__ h1, const float* __restrict__ cvec,
                             float* __restrict__ alpha) {
    int pid  = blockIdx.x * 8 + (threadIdx.x >> 5);
    int lane = threadIdx.x & 31;
    if (pid >= cNP) return;
    int gg = pid >> 7;
    float4 a = ((const float4*)(h1 + (size_t)pid * cH))[lane];
    float4 b = ((const float4*)(cvec + (size_t)gg * cH))[lane];
    float acc = a.x * b.x + a.y * b.y + a.z * b.z + a.w * b.w;
#pragma unroll
    for (int off = 16; off > 0; off >>= 1) acc += __shfl_down(acc, off, 32);
    if (lane == 0) alpha[pid] = 1.0f / (1.0f + __expf(-acc));
}

__global__ void pvec_kernel(const float* __restrict__ h1, const float* __restrict__ alpha,
                            float* __restrict__ pvec) {
    int gg = blockIdx.x;
    int c  = threadIdx.x;
    float acc = 0.0f;
    for (int i = 0; i < cK; ++i) {
        int pid = gg * cK + i;
        acc += alpha[pid] * h1[(size_t)pid * cH + c];
    }
    pvec[gg * cH + c] = acc;
}

// =====================================================================
// Final MLP: concat(p1,p2)[256] -> 128 -> 64 -> 10   (one block per graph)
// =====================================================================
__global__ void mlp_kernel(const float* __restrict__ pvec,
                           const float* __restrict__ W1, const float* __restrict__ b1,
                           const float* __restrict__ W2, const float* __restrict__ b2,
                           const float* __restrict__ W3, const float* __restrict__ b3,
                           float* __restrict__ out) {
    __shared__ float pv[256];
    __shared__ float a1[128];
    __shared__ float a2[64];
    int g = blockIdx.x;     // 0..127
    int t = threadIdx.x;    // 0..255
    if (t < 128) pv[t] = pvec[g * cH + t];
    else         pv[t] = pvec[(cB + g) * cH + (t - 128)];
    __syncthreads();
    if (t < 128) {
        float acc = b1[t];
        for (int i = 0; i < 256; ++i) acc += pv[i] * W1[i * 128 + t];
        a1[t] = fmaxf(acc, 0.0f);
    }
    __syncthreads();
    if (t < 64) {
        float acc = b2[t];
        for (int i = 0; i < 128; ++i) acc += a1[i] * W2[i * 64 + t];
        a2[t] = fmaxf(acc, 0.0f);
    }
    __syncthreads();
    if (t < 10) {
        float acc = b3[t];
        for (int i = 0; i < 64; ++i) acc += a2[i] * W3[i * 10 + t];
        out[cN + g * 10 + t] = acc;
    }
}

// =====================================================================
// Host launcher
// =====================================================================
extern "C" void kernel_launch(void* const* d_in, const int* in_sizes, int n_in,
                              void* d_out, int out_size, void* d_ws, size_t ws_size,
                              hipStream_t stream) {
    const float* x       = (const float*)d_in[0];
    const int*   ei      = (const int*)d_in[1];
    const float* conv_W[3] = { (const float*)d_in[2], (const float*)d_in[4], (const float*)d_in[6] };
    const float* conv_b[3] = { (const float*)d_in[3], (const float*)d_in[5], (const float*)d_in[7] };
    const float* pool_Wr = (const float*)d_in[8];
    const float* pool_Ws = (const float*)d_in[9];
    const float* pool_b  = (const float*)d_in[10];
    const float* final_W = (const float*)d_in[11];
    const float* final_b = (const float*)d_in[12];
    const float* gp_W    = (const float*)d_in[13];
    const float* lin1_W  = (const float*)d_in[14];
    const float* lin1_b  = (const float*)d_in[15];
    const float* lin2_W  = (const float*)d_in[16];
    const float* lin2_b  = (const float*)d_in[17];
    const float* lin3_W  = (const float*)d_in[18];
    const float* lin3_b  = (const float*)d_in[19];
    float* out = (float*)d_out;

    const int* esrc = ei;
    const int* edst = ei + cE;

    // ---- workspace layout (floats) ----
    float*  W  = (float*)d_ws;
    size_t  o  = 0;
    float* dinv  = W + o; o += cN;
    float* hcat  = W + o; o += (size_t)cN * cD;
    float* xw    = W + o; o += (size_t)cN * cH;     // reused as xwp (cNP*cH)
    float* agg   = W + o; o += (size_t)cN * cH;     // reused: aggp + h1buf
    float* xp    = W + o; o += (size_t)cNP * cD;
    float* yr    = W + o; o += cN;
    float* ys    = W + o; o += cN;
    float* sacc  = W + o; o += cN;                  // also temp degree buffer
    float* score = W + o; o += cN;
    int*   keep  = (int*)(W + o); o += cN;
    int*   inv   = (int*)(W + o); o += cNP;
    float* dinvp = W + o; o += cNP;                 // degp then rsqrt in-place
    float* xsum  = W + o; o += (size_t)cNG * cH;
    float* cvec  = W + o; o += (size_t)cNG * cH;
    float* alpha = W + o; o += cNP;
    float* pvec  = W + o; o += (size_t)cNG * cH;
    (void)ws_size; (void)n_in; (void)in_sizes; (void)out_size;

    float* xwp   = xw;                 // cNP x cH
    float* aggp  = agg;                // cNP x cH
    float* h1buf = agg + (size_t)cNP * cH;

    const int T = 256;
    const size_t ldsH = (size_t)cH * 128 * sizeof(bf16_t);   // 32 KB
    const size_t ldsD = (size_t)cD * 128 * sizeof(bf16_t);   // 96 KB

    // batch_ output (constant ramp)
    write_batch<<<cN / T, T, 0, stream>>>(out);

    // degrees -> dinv (all edge weights = 1)
    fill_zero<<<cN / T, T, 0, stream>>>(sacc, cN);
    edge_count<<<cE / T, T, 0, stream>>>(edst, sacc);
    rsqrt_deg<<<cN / T, T, 0, stream>>>(sacc, dinv, cN);

    // 3 GCN layers -> hcat[:, l*128 : (l+1)*128]
    for (int l = 0; l < 3; ++l) {
        const float* Ain = (l == 0) ? x : (hcat + (size_t)(l - 1) * cH);
        int lda = (l == 0) ? cH : cD;
        wmma_gemm<<<cN / 128, T, ldsH, stream>>>(Ain, lda, conv_W[l], xw, cH, cH);
        fill_zero<<<(cN * cH) / T, T, 0, stream>>>(agg, cN * cH);
        gcn_agg<<<cE / 8, T, 0, stream>>>(esrc, edst, dinv, xw, agg);
        gcn_finish<<<(cN * cH) / T, T, 0, stream>>>(agg, xw, dinv, conv_b[l],
                                                    hcat + (size_t)l * cH, cD, cN);
    }

    // SAGPool scorer (linearity: project to scalar first, then edge scatter)
    score_dots<<<cN / 8, T, 0, stream>>>(hcat, pool_Wr, pool_Ws, yr, ys);
    fill_zero<<<cN / T, T, 0, stream>>>(sacc, cN);
    edge_score<<<cE / T, T, 0, stream>>>(esrc, edst, yr, sacc);
    score_finish<<<cN / T, T, 0, stream>>>(sacc, ys, pool_b, score);

    // top-K + gather pooled features
    topk_kernel<<<cNG, 256, 0, stream>>>(score, keep, inv);
    pool_gather<<<(cNP * cD) / T, T, 0, stream>>>(hcat, score, inv, xp);

    // pooled degrees -> dinvp
    fill_zero<<<cNP / T, T, 0, stream>>>(dinvp, cNP);
    pooled_deg<<<cE / T, T, 0, stream>>>(esrc, edst, keep, dinvp);
    rsqrt_deg<<<cNP / T, T, 0, stream>>>(dinvp, dinvp, cNP);

    // pooled GCN conv (D=384 -> 128) via WMMA
    wmma_gemm<<<cNP / 128, T, ldsD, stream>>>(xp, cD, final_W, xwp, cH, cD);
    fill_zero<<<(cNP * cH) / T, T, 0, stream>>>(aggp, cNP * cH);
    pooled_agg<<<cE / 8, T, 0, stream>>>(esrc, edst, keep, dinvp, xwp, aggp);
    gcn_finish<<<(cNP * cH) / T, T, 0, stream>>>(aggp, xwp, dinvp, final_b, h1buf, cH, cNP);

    // attention pooling over both partitions (256 pooled graphs)
    xsum_kernel<<<cNG, cH, 0, stream>>>(h1buf, xsum);
    cvec_kernel<<<(cNG * cH) / T, T, 0, stream>>>(xsum, gp_W, cvec);
    alpha_kernel<<<cNP / 8, T, 0, stream>>>(h1buf, cvec, alpha);
    pvec_kernel<<<cNG, cH, 0, stream>>>(h1buf, alpha, pvec);

    // final MLP -> pred at out[cN ..]
    mlp_kernel<<<cB, 256, 0, stream>>>(pvec, lin1_W, lin1_b, lin2_W, lin2_b,
                                       lin3_W, lin3_b, out);
}
